// SAGEDepth_80676665688568
// MI455X (gfx1250) — compile-verified, matrix operand order checked by reading, not run on hardware
//
#include <hip/hip_runtime.h>

// CDNA5 / gfx1250. wave32. WMMA bf16 16x16x32 with f32 accumulation.

typedef __attribute__((ext_vector_type(16))) __bf16 v16bf;
typedef __attribute__((ext_vector_type(8)))  float  v8f;

#define CIN  128
#define CHID 256
#define COUT 64
#define EPSB 1e-5f

__device__ __forceinline__ unsigned short f32_to_bf16(float f) {
  unsigned int x = __float_as_uint(f);
  x += 0x7FFFu + ((x >> 16) & 1u);   // round-to-nearest-even
  return (unsigned short)(x >> 16);
}
__device__ __forceinline__ float bf16_to_f32(unsigned short h) {
  return __uint_as_float(((unsigned int)h) << 16);
}

// ---------------- utility kernels ----------------

__global__ void k_zero(uint4* __restrict__ p, long long n4) {
  long long i = blockIdx.x * (long long)blockDim.x + threadIdx.x;
  if (i < n4) p[i] = make_uint4(0u, 0u, 0u, 0u);
}

__global__ void k_count(const long long* __restrict__ dst, float* __restrict__ cnt, int E) {
  int e = blockIdx.x * blockDim.x + threadIdx.x;
  if (e < E) {
    int d = (int)dst[e];
    __hip_atomic_fetch_add(&cnt[d], 1.0f, __ATOMIC_RELAXED, __HIP_MEMORY_SCOPE_AGENT);
  }
}

__global__ void k_inv(const float* __restrict__ cnt, float* __restrict__ inv, int n) {
  int i = blockIdx.x * blockDim.x + threadIdx.x;
  if (i < n) inv[i] = 1.0f / fmaxf(cnt[i], 1.0f);
}

// scale rows by inv[row] (or 1.0 if inv==nullptr), convert f32 -> bf16; pad rows -> 0
// cShift = log2(C/4)
__global__ void k_scale_cvt(const float* __restrict__ in, const float* __restrict__ inv,
                            unsigned short* __restrict__ out,
                            int cShift, int Nreal, long long total4) {
  long long t = blockIdx.x * (long long)blockDim.x + threadIdx.x;
  if (t >= total4) return;
  long long row = t >> cShift;
  int c = (int)(t & ((1 << cShift) - 1)) << 2;
  size_t off = ((size_t)row << (cShift + 2)) + c;   // row*C + c
  ushort4 o;
  if (row < Nreal) {
    float4 v = *reinterpret_cast<const float4*>(in + off);
    float s = inv ? inv[row] : 1.0f;
    o.x = f32_to_bf16(v.x * s);
    o.y = f32_to_bf16(v.y * s);
    o.z = f32_to_bf16(v.z * s);
    o.w = f32_to_bf16(v.w * s);
  } else {
    o = make_ushort4(0, 0, 0, 0);
  }
  *reinterpret_cast<ushort4*>(out + off) = o;
}

// one thread per (edge, 8 channels): bf16 gather + 8 f32 atomic adds (L2-resident)
// cShift = log2(C/8). For C=256 one wave == one edge -> fully coalesced 512B gather.
__global__ void k_aggregate(const long long* __restrict__ src,
                            const long long* __restrict__ dst,
                            const unsigned short* __restrict__ feat,
                            float* __restrict__ agg,
                            int cShift, long long total8) {
  long long t = blockIdx.x * (long long)blockDim.x + threadIdx.x;
  if (t >= total8) return;
  long long e = t >> cShift;
  int c = (int)(t & ((1 << cShift) - 1)) << 3;
  int C = 8 << cShift;
  int s = (int)src[e];
  int d = (int)dst[e];
  union { uint4 q; unsigned short h[8]; } u;
  u.q = *reinterpret_cast<const uint4*>(feat + (size_t)s * C + c);
  float* o = agg + (size_t)d * C + c;
#pragma unroll
  for (int i = 0; i < 8; ++i)
    __hip_atomic_fetch_add(o + i, bf16_to_f32(u.h[i]),
                           __ATOMIC_RELAXED, __HIP_MEMORY_SCOPE_AGENT);
}

// Pre-swizzle W (K x Nout, f32, row-major) into the per-lane B-fragment layout of
// V_WMMA_F32_16X16X32_BF16. Tile (kt,nt) occupies 512 bf16 = 32 lanes x 16 elems:
//   lane l: N = nt*16 + (l&15), khalf = l>>4
//   elem i: K = kt*32 + khalf*8 + (i<8 ? i : 16 + (i-8))
__global__ void k_pack_b(const float* __restrict__ W, unsigned short* __restrict__ out,
                         int K, int Nout) {
  int tid = blockIdx.x * blockDim.x + threadIdx.x;
  int total = K * Nout;
  if (tid >= total) return;
  int i    = tid & 15;
  int lane = (tid >> 4) & 31;
  int tile = tid >> 9;
  int ntTotal = Nout >> 4;
  int kt = tile / ntTotal;
  int nt = tile - kt * ntTotal;
  int khalf = lane >> 4;
  int n = nt * 16 + (lane & 15);
  int k = kt * 32 + khalf * 8 + (i < 8 ? i : 16 + (i - 8));
  out[tid] = f32_to_bf16(W[(size_t)k * Nout + n]);
}

// ---------------- fused dual-GEMM + bias (+BN+ReLU) ----------------
// out[row, :] = agg[row,:] @ Wl + bias + root[row,:] @ Wr   (then BN+ReLU if DO_BN)
// Block: 256 threads = 8 waves; covers 64 rows x NOUT cols.
// Per-wave tile: MT row-tiles x NTW col-tiles (16x16 each).
//   row groups RGC = 4/MT (each spans MT*16 rows), col groups CGC = 8/RGC.
// All fragment loads for one k-step are hoisted into register arrays so they
// issue as one clause and WMMAs drain loadcnt incrementally.
// K = KT*32; every fragment address is base + immediate offset.
// Rows padded to multiple of 64 (EXEC all-ones for WMMA).
template <int KT, int MT, int NOUT, bool DO_BN, bool OUT_F32>
__global__ __launch_bounds__(256) void k_sage_gemm(
    const unsigned short* __restrict__ Aagg,   // bf16, Npad x K
    const unsigned short* __restrict__ Aroot,  // bf16, Npad x K
    const unsigned short* __restrict__ Bl,     // packed bf16 panel
    const unsigned short* __restrict__ Br,     // packed bf16 panel
    const float* __restrict__ bias,
    const float* __restrict__ g,  const float* __restrict__ be,
    const float* __restrict__ rm, const float* __restrict__ rv,
    float* __restrict__ outF, unsigned short* __restrict__ outB,
    int Nreal)
{
  constexpr int K   = KT * 32;
  constexpr int NTT = NOUT / 16;        // total 16-col tiles
  constexpr int RGC = 4 / MT;           // row groups per block
  constexpr int CGC = 8 / RGC;          // col groups per block
  constexpr int NTW = NTT / CGC;        // col tiles per wave
  constexpr size_t B_KT_STRIDE = (size_t)NTT * 1024;   // bytes per kt step
  constexpr size_t A_MT_STRIDE = (size_t)16 * K * 2;   // bytes per row tile

  const int tid   = threadIdx.x;
  const int wid   = tid >> 5;
  const int lane  = tid & 31;
  const int rg    = wid % RGC;
  const int cg    = wid / RGC;
  const int rowBase = blockIdx.x * 64 + rg * (MT * 16);
  const int m     = lane & 15;
  const int khalf = lane >> 4;

  union Frag { v16bf v; uint4 q[2]; };

  v8f acc[MT][NTW];
  v8f zero = {};
#pragma unroll
  for (int mt = 0; mt < MT; ++mt)
#pragma unroll
    for (int t = 0; t < NTW; ++t) acc[mt][t] = zero;

  const size_t aLaneOff = ((size_t)(rowBase + m) * K + khalf * 8) * 2;
  const size_t bLaneOff = (((size_t)cg * NTW) * 32 + lane) * 32;

#pragma unroll 1
  for (int pass = 0; pass < 2; ++pass) {
    const char* aBase = (const char*)(pass ? Aroot : Aagg) + aLaneOff;
    const char* bBase = (const char*)(pass ? Br : Bl) + bLaneOff;
#pragma unroll 2
    for (int kt = 0; kt < KT; ++kt) {
      Frag af[MT], bf[NTW];
#pragma unroll
      for (int mt = 0; mt < MT; ++mt) {
        af[mt].q[0] = *reinterpret_cast<const uint4*>(aBase + mt * A_MT_STRIDE);
        af[mt].q[1] = *reinterpret_cast<const uint4*>(aBase + mt * A_MT_STRIDE + 32);
      }
#pragma unroll
      for (int t = 0; t < NTW; ++t) {
        bf[t].q[0] = *reinterpret_cast<const uint4*>(bBase + t * 1024);
        bf[t].q[1] = *reinterpret_cast<const uint4*>(bBase + t * 1024 + 16);
      }
#pragma unroll
      for (int mt = 0; mt < MT; ++mt)
#pragma unroll
        for (int t = 0; t < NTW; ++t)
          acc[mt][t] = __builtin_amdgcn_wmma_f32_16x16x32_bf16(
              false, af[mt].v, false, bf[t].v, (short)0, acc[mt][t], false, false);
      aBase += 64;
      bBase += B_KT_STRIDE;
    }
  }

  // Epilogue. C/D layout: VGPR v, lanes 0-15 -> M=v, lanes 16-31 -> M=v+8; N=lane&15.
#pragma unroll
  for (int mt = 0; mt < MT; ++mt) {
#pragma unroll
    for (int t = 0; t < NTW; ++t) {
      const int col = cg * (NTW * 16) + t * 16 + (lane & 15);
      float scale = 1.0f, shift = bias[col];
      if (DO_BN) {
        float s = g[col] * rsqrtf(rv[col] + EPSB);
        scale = s;
        shift = (bias[col] - rm[col]) * s + be[col];
      }
#pragma unroll
      for (int v = 0; v < 8; ++v) {
        int row = rowBase + mt * 16 + (khalf << 3) + v;
        float val = acc[mt][t][v] * scale + shift;
        if (DO_BN) val = fmaxf(val, 0.0f);
        if (OUT_F32) {
          if (row < Nreal) outF[(size_t)row * NOUT + col] = val;
        } else {
          outB[(size_t)row * NOUT + col] = f32_to_bf16(val);
        }
      }
    }
  }
}

// ---------------- host-side orchestration ----------------

static inline long long cdiv(long long a, long long b) { return (a + b - 1) / b; }

extern "C" void kernel_launch(void* const* d_in, const int* in_sizes, int n_in,
                              void* d_out, int out_size, void* d_ws, size_t ws_size,
                              hipStream_t stream) {
  const float*     x    = (const float*)d_in[0];
  const long long* eidx = (const long long*)d_in[1];   // int64 (2, E)
  const float* Wl0 = (const float*)d_in[2];  const float* bl0 = (const float*)d_in[3];
  const float* Wr0 = (const float*)d_in[4];
  const float* Wl1 = (const float*)d_in[5];  const float* bl1 = (const float*)d_in[6];
  const float* Wr1 = (const float*)d_in[7];
  const float* Wl2 = (const float*)d_in[8];  const float* bl2 = (const float*)d_in[9];
  const float* Wr2 = (const float*)d_in[10];
  const float* g0  = (const float*)d_in[11]; const float* be0 = (const float*)d_in[12];
  const float* rm0 = (const float*)d_in[13]; const float* rv0 = (const float*)d_in[14];
  const float* g1  = (const float*)d_in[15]; const float* be1 = (const float*)d_in[16];
  const float* rm1 = (const float*)d_in[17]; const float* rv1 = (const float*)d_in[18];

  const int N = in_sizes[0] / CIN;
  const int E = in_sizes[1] / 2;
  const long long Npad = cdiv(N, 64) * 64;
  const long long* esrc = eidx;
  const long long* edst = eidx + E;

  // workspace carve-out (256B aligned)
  size_t off = 0;
  auto carve = [&](size_t bytes) -> void* {
    void* p = (char*)d_ws + off;
    off = (off + bytes + 255) & ~(size_t)255;
    return p;
  };
  unsigned short* xB   = (unsigned short*)carve(Npad * CIN * 2);   // x in bf16
  unsigned short* hA   = (unsigned short*)carve(Npad * CHID * 2);  // hidden 0
  unsigned short* hB   = (unsigned short*)carve(Npad * CHID * 2);  // hidden 1
  float*          aggF = (float*)carve(Npad * CHID * 4);           // f32 accumulators
  unsigned short* aggB = (unsigned short*)carve(Npad * CHID * 2);  // scaled bf16 agg
  float*          cnt  = (float*)carve(Npad * 4);
  float*          inv  = (float*)carve(Npad * 4);
  unsigned short* Wl0p = (unsigned short*)carve(CIN * CHID * 2);
  unsigned short* Wr0p = (unsigned short*)carve(CIN * CHID * 2);
  unsigned short* Wl1p = (unsigned short*)carve(CHID * CHID * 2);
  unsigned short* Wr1p = (unsigned short*)carve(CHID * CHID * 2);
  unsigned short* Wl2p = (unsigned short*)carve(CHID * COUT * 2);
  unsigned short* Wr2p = (unsigned short*)carve(CHID * COUT * 2);
  (void)ws_size; (void)n_in; (void)out_size;

  const int T = 256;

  // 1) degrees -> 1/max(deg,1)
  k_zero<<<(int)cdiv(Npad / 4, T), T, 0, stream>>>((uint4*)cnt, Npad / 4);
  k_count<<<(int)cdiv(E, T), T, 0, stream>>>(edst, cnt, E);
  k_inv<<<(int)cdiv(Npad, T), T, 0, stream>>>(cnt, inv, (int)Npad);

  // 2) x -> bf16 (padded rows zeroed)
  {
    long long t4 = Npad * (CIN / 4);
    k_scale_cvt<<<(int)cdiv(t4, T), T, 0, stream>>>(x, nullptr, xB, 5, N, t4);
  }

  // 3) pack all six weight panels into WMMA B layout
  k_pack_b<<<cdiv(CIN * CHID, T), T, 0, stream>>>(Wl0, Wl0p, CIN, CHID);
  k_pack_b<<<cdiv(CIN * CHID, T), T, 0, stream>>>(Wr0, Wr0p, CIN, CHID);
  k_pack_b<<<cdiv(CHID * CHID, T), T, 0, stream>>>(Wl1, Wl1p, CHID, CHID);
  k_pack_b<<<cdiv(CHID * CHID, T), T, 0, stream>>>(Wr1, Wr1p, CHID, CHID);
  k_pack_b<<<cdiv(CHID * COUT, T), T, 0, stream>>>(Wl2, Wl2p, CHID, COUT);
  k_pack_b<<<cdiv(CHID * COUT, T), T, 0, stream>>>(Wr2, Wr2p, CHID, COUT);

  const int gemmGrid = (int)(Npad / 64);

  // ---- layer 0: C_IN=128 -> C_HID=256, BN+ReLU, bf16 out (hA) ----
  k_zero<<<(int)cdiv(Npad * CIN / 4, T), T, 0, stream>>>((uint4*)aggF, Npad * CIN / 4);
  {
    long long t8 = (long long)E * (CIN / 8);
    k_aggregate<<<(int)cdiv(t8, T), T, 0, stream>>>(esrc, edst, xB, aggF, 4, t8);
    long long t4 = Npad * (CIN / 4);
    k_scale_cvt<<<(int)cdiv(t4, T), T, 0, stream>>>(aggF, inv, aggB, 5, N, t4);
  }
  k_sage_gemm<4, 2, CHID, true, false><<<gemmGrid, 256, 0, stream>>>(
      aggB, xB, Wl0p, Wr0p, bl0, g0, be0, rm0, rv0,
      nullptr, hA, N);

  // ---- layer 1: C_HID -> C_HID, BN+ReLU, bf16 out (hB) ----
  k_zero<<<(int)cdiv(Npad * CHID / 4, T), T, 0, stream>>>((uint4*)aggF, Npad * CHID / 4);
  {
    long long t8 = (long long)E * (CHID / 8);
    k_aggregate<<<(int)cdiv(t8, T), T, 0, stream>>>(esrc, edst, hA, aggF, 5, t8);
    long long t4 = Npad * (CHID / 4);
    k_scale_cvt<<<(int)cdiv(t4, T), T, 0, stream>>>(aggF, inv, aggB, 6, N, t4);
  }
  k_sage_gemm<8, 2, CHID, true, false><<<gemmGrid, 256, 0, stream>>>(
      aggB, hA, Wl1p, Wr1p, bl1, g1, be1, rm1, rv1,
      nullptr, hB, N);

  // ---- layer 2: C_HID -> C_OUT, no BN, f32 out -> d_out ----
  k_zero<<<(int)cdiv(Npad * CHID / 4, T), T, 0, stream>>>((uint4*)aggF, Npad * CHID / 4);
  {
    long long t8 = (long long)E * (CHID / 8);
    k_aggregate<<<(int)cdiv(t8, T), T, 0, stream>>>(esrc, edst, hB, aggF, 5, t8);
    long long t4 = Npad * (CHID / 4);
    k_scale_cvt<<<(int)cdiv(t4, T), T, 0, stream>>>(aggF, inv, aggB, 6, N, t4);
  }
  k_sage_gemm<8, 1, COUT, false, true><<<gemmGrid, 256, 0, stream>>>(
      aggB, hB, Wl2p, Wr2p, bl2, nullptr, nullptr, nullptr, nullptr,
      (float*)d_out, nullptr, N);
}
